// Head_5162550690269
// MI455X (gfx1250) — matrix operand order repaired and checked by reading, not compiled, not run on hardware
//
#include <hip/hip_runtime.h>

typedef __attribute__((ext_vector_type(16))) _Float16     v16h;
typedef __attribute__((ext_vector_type(8)))  float        v8f;
typedef __attribute__((ext_vector_type(4)))  unsigned int v4u;
typedef __attribute__((ext_vector_type(8)))  int          v8i;
typedef __attribute__((ext_vector_type(4)))  int          v4i;

#define EMB 1024
#define HS  64
#define BB  4
#define TT  4096
#define BT  (BB*TT)

// A-fragment K offset inside a 32-wide chunk (16-bit A layout, ISA 7.12.2):
// lanes 0-15 hold K {0..7,16..23}, lanes 16-31 hold K {8..15,24..31}
__device__ __forceinline__ int ka_idx(int j, int half) {
    return (j < 8 ? j : j + 8) + 8 * half;
}

// TDM: DMA a 64x64 f16 tile (row stride 64 elements) from global -> LDS.
// D# packing per ISA 8.3/8.4: group0 {count=1, lds_addr, global_addr, type=2},
// group1 {data_size=1(2B), tensor_dim0=64, tensor_dim1=64, tile_dim0=64,
// tile_dim1=64, dim0_stride=64, dim1_stride=4096}; trailing groups zero (2-D).
__device__ __forceinline__ void tdm_load_tile_64x64_f16(unsigned lds_off,
                                                        const _Float16* gptr)
{
    unsigned long long ga = (unsigned long long)(uintptr_t)gptr;
    v4u g0;
    g0[0] = 1u;                                            // count=1, user D#
    g0[1] = lds_off;                                       // lds_addr (bytes)
    g0[2] = (unsigned)(ga & 0xffffffffu);                  // global_addr[31:0]
    g0[3] = (unsigned)((ga >> 32) & 0x1ffffffu) | (2u << 30); // GA[56:32]|type=2
    v8i g1;
    g1[0] = (int)(1u << 16);        // workgroup_mask=0, data_size=1 (2 bytes)
    g1[1] = (int)(64u << 16);       // tensor_dim0[15:0]=64 (D# bits 63:48 of grp)
    g1[2] = (int)(64u << 16);       // tensor_dim0[31:16]=0, tensor_dim1[15:0]=64
    g1[3] = (int)(64u << 16);       // tensor_dim1[31:16]=0, tile_dim0=64
    g1[4] = (int)64u;               // tile_dim1=64, tile_dim2=0 (2-D)
    g1[5] = (int)64u;               // tensor_dim0_stride[31:0]=64
    g1[6] = (int)(4096u << 16);     // d0_stride[47:32]=0, d1_stride[15:0]=4096
    g1[7] = 0;                      // tensor_dim1_stride[47:16]=0
    v4i gz4 = {0, 0, 0, 0};
    v8i gz8 = {0, 0, 0, 0, 0, 0, 0, 0};
    __builtin_amdgcn_tensor_load_to_lds(g0, g1, gz4, gz4, gz8, 0);
}

// ---------------- Stage 1: fused Q/K/V projection  (x @ W), f32 -> f16 ----
// grid = (BT/16, 3), block = 128 (4 waves; wave w owns output cols [16w,16w+16))
__global__ __launch_bounds__(128) void proj_qkv_kernel(
    const float* __restrict__ x,
    const float* __restrict__ Wq,
    const float* __restrict__ Wk,
    const float* __restrict__ Wv,
    _Float16* __restrict__ Qh,
    _Float16* __restrict__ Kh,
    _Float16* __restrict__ Vh)
{
    const int lane = threadIdx.x & 31;
    const int wave = threadIdx.x >> 5;
    const int half = lane >> 4;
    const int l16  = lane & 15;

    const int row0  = blockIdx.x * 16;   // token rows
    const int n0    = wave * 16;         // head-dim cols
    const int which = blockIdx.y;        // 0=Q 1=K 2=V

    const float* W = (which == 0) ? Wq : (which == 1) ? Wk : Wv;
    _Float16* Out  = (which == 0) ? Qh : (which == 1) ? Kh : Vh;

    const float* xrow = x + (size_t)(row0 + l16) * EMB;

    v8f acc;
    #pragma unroll
    for (int i = 0; i < 8; ++i) acc[i] = 0.0f;

    for (int kb = 0; kb < EMB; kb += 32) {
        v16h a;   // 16x32 slice of x, f32 -> f16, two b128 pairs per lane
        {
            const int g0 = kb + 16 * half;
            const float4 f0 = *(const float4*)(xrow + g0);
            const float4 f1 = *(const float4*)(xrow + g0 + 4);
            const float4 f2 = *(const float4*)(xrow + g0 + 8);
            const float4 f3 = *(const float4*)(xrow + g0 + 12);
            a[0]=(_Float16)f0.x;  a[1]=(_Float16)f0.y;  a[2]=(_Float16)f0.z;  a[3]=(_Float16)f0.w;
            a[4]=(_Float16)f1.x;  a[5]=(_Float16)f1.y;  a[6]=(_Float16)f1.z;  a[7]=(_Float16)f1.w;
            a[8]=(_Float16)f2.x;  a[9]=(_Float16)f2.y;  a[10]=(_Float16)f2.z; a[11]=(_Float16)f2.w;
            a[12]=(_Float16)f3.x; a[13]=(_Float16)f3.y; a[14]=(_Float16)f3.z; a[15]=(_Float16)f3.w;
        }
        v16h b;   // 32x16 slice of W; per-lane column, contiguous K rows
        {
            const float* wp = W + (size_t)(kb + 16 * half) * HS + n0 + l16;
            #pragma unroll
            for (int j = 0; j < 16; ++j) b[j] = (_Float16)wp[j * HS];
        }
        acc = __builtin_amdgcn_wmma_f32_16x16x32_f16(false, a, false, b,
                                                     (short)0, acc, false, false);
    }

    // fold emb_dim^-0.5 (=1/32) into Q so attention needs no score scaling
    const float qscale = (which == 0) ? 0.03125f : 1.0f;
    #pragma unroll
    for (int r = 0; r < 8; ++r) {
        const int row = row0 + r + 8 * half;
        Out[(size_t)row * HS + n0 + l16] = (_Float16)(acc[r] * qscale);
    }
}

// ------- Stage 2: causal flash attention, 4 waves / 64 queries per block ----
// K/V tiles staged to LDS by the Tensor Data Mover, double-buffered.
// grid = (TT/64, BB), block = 128
__global__ __launch_bounds__(128) void flash_attn_kernel(
    const _Float16* __restrict__ Qh,
    const _Float16* __restrict__ Kh,
    const _Float16* __restrict__ Vh,
    float* __restrict__ out)
{
    __shared__ _Float16 KT[2][64 * 64];     // 2 x 8 KB key tiles
    __shared__ _Float16 VT[2][64 * 64];     // 2 x 8 KB value tiles
    __shared__ _Float16 Ptile[4][16 * 64];  // per-wave C->A layout staging

    const int lane = threadIdx.x & 31;
    const int wave = threadIdx.x >> 5;
    const int half = lane >> 4;
    const int l16  = lane & 15;
    const int b    = blockIdx.y;
    const int q0b  = blockIdx.x * 64;       // block's first query
    const int q0   = q0b + wave * 16;       // this wave's first query

    const _Float16* Qb = Qh + (size_t)b * TT * HS;
    const _Float16* Kb = Kh + (size_t)b * TT * HS;
    const _Float16* Vb = Vh + (size_t)b * TT * HS;
    _Float16* Pw = &Ptile[wave][0];

    // Q A-fragments (K=64 -> two 32-chunks), resident for the whole kernel
    v16h aq[2];
    #pragma unroll
    for (int c = 0; c < 2; ++c) {
        const _Float16* qp = Qb + (size_t)(q0 + l16) * HS + c * 32;
        #pragma unroll
        for (int j = 0; j < 16; ++j) aq[c][j] = qp[ka_idx(j, half)];
    }

    // Online-softmax state: slot r <-> query row (q0 + r + 8*half), C layout
    float m_i[8], l_i[8];
    v8f o[4];
    #pragma unroll
    for (int r = 0; r < 8; ++r) { m_i[r] = -__builtin_inff(); l_i[r] = 0.0f; }
    #pragma unroll
    for (int t = 0; t < 4; ++t)
        #pragma unroll
        for (int r = 0; r < 8; ++r) o[t][r] = 0.0f;

    const float LOG2E = 1.44269504088896340736f;
    const int ntiles = q0b / 64 + 1;        // causal: keys [0, q0b+64)

    // Prologue: TDM tile 0 into buffer 0
    if (wave == 0) {
        tdm_load_tile_64x64_f16((unsigned)(size_t)&KT[0][0], Kb);
        tdm_load_tile_64x64_f16((unsigned)(size_t)&VT[0][0], Vb);
    }
    __builtin_amdgcn_s_wait_tensorcnt(0);   // no-op for waves 1..3
    __syncthreads();

    for (int it = 0; it < ntiles; ++it) {
        const int k0  = it * 64;
        const int cur = it & 1;
        const int nxt = cur ^ 1;

        // Kick off the next tile's DMA while we compute on the current one
        if (wave == 0 && (it + 1) < ntiles) {
            tdm_load_tile_64x64_f16((unsigned)(size_t)&KT[nxt][0],
                                    Kb + (size_t)(k0 + 64) * HS);
            tdm_load_tile_64x64_f16((unsigned)(size_t)&VT[nxt][0],
                                    Vb + (size_t)(k0 + 64) * HS);
        }

        if (k0 <= q0 + 15) {   // this wave has non-masked work in this tile
            // ---- scores S = Q.K^T for a 16x64 tile (4 C-fragments) ----
            v8f sf[4];
            #pragma unroll
            for (int nb = 0; nb < 4; ++nb) {
                const int kc = k0 + nb * 16;
                if (kc >= q0 + 16) {          // entirely in the future
                    #pragma unroll
                    for (int r = 0; r < 8; ++r) sf[nb][r] = -__builtin_inff();
                    continue;
                }
                v8f s;
                #pragma unroll
                for (int r = 0; r < 8; ++r) s[r] = 0.0f;
                #pragma unroll
                for (int c = 0; c < 2; ++c) {
                    // B fragment of K^T from LDS: element (h,n) = K[n][h]
                    v16h bk;
                    const _Float16* kp =
                        &KT[cur][(nb * 16 + l16) * 64 + c * 32];
                    #pragma unroll
                    for (int j = 0; j < 16; ++j) bk[j] = kp[ka_idx(j, half)];
                    s = __builtin_amdgcn_wmma_f32_16x16x32_f16(
                            false, aq[c], false, bk, (short)0, s, false, false);
                }
                #pragma unroll
                for (int r = 0; r < 8; ++r) {   // causal mask in C layout
                    const int qrow = q0 + r + 8 * half;
                    const int kcol = kc + l16;
                    sf[nb][r] = (kcol > qrow) ? -__builtin_inff() : s[r];
                }
            }

            // ---- row max (over nb, then over the 16-lane half) ----
            float mnew[8], alpha[8], ts[8];
            #pragma unroll
            for (int r = 0; r < 8; ++r) {
                float v = fmaxf(fmaxf(sf[0][r], sf[1][r]),
                                fmaxf(sf[2][r], sf[3][r]));
                #pragma unroll
                for (int m = 1; m < 16; m <<= 1) v = fmaxf(v, __shfl_xor(v, m, 32));
                mnew[r]  = fmaxf(m_i[r], v);    // finite: diagonal always valid
                alpha[r] = __builtin_exp2f((m_i[r] - mnew[r]) * LOG2E);
                m_i[r]   = mnew[r];
                ts[r]    = 0.0f;
            }

            // ---- P = exp(S - m); stage to this wave's LDS tile (f16) ----
            #pragma unroll
            for (int nb = 0; nb < 4; ++nb) {
                #pragma unroll
                for (int r = 0; r < 8; ++r) {
                    const float p = __builtin_exp2f((sf[nb][r] - mnew[r]) * LOG2E);
                    ts[r] += p;
                    Pw[(r + 8 * half) * 64 + nb * 16 + l16] = (_Float16)p;
                }
            }
            #pragma unroll
            for (int r = 0; r < 8; ++r) {
                float v = ts[r];
                #pragma unroll
                for (int m = 1; m < 16; m <<= 1) v += __shfl_xor(v, m, 32);
                l_i[r] = l_i[r] * alpha[r] + v;
            }
            #pragma unroll
            for (int t = 0; t < 4; ++t)
                #pragma unroll
                for (int r = 0; r < 8; ++r) o[t][r] *= alpha[r];

            // Wave-private LDS RAW fence (LDS is in-order per wave + dscnt wait)
            asm volatile("s_wait_dscnt 0x0" ::: "memory");

            // ---- O += P @ V  (2 K-chunks x 4 head-dim tiles), V from LDS ----
            #pragma unroll
            for (int c = 0; c < 2; ++c) {
                v16h ap;
                const _Float16* pp = Pw + l16 * 64 + c * 32;
                #pragma unroll
                for (int j = 0; j < 16; ++j) ap[j] = pp[ka_idx(j, half)];
                #pragma unroll
                for (int t = 0; t < 4; ++t) {
                    v16h bv;
                    const _Float16* vp =
                        &VT[cur][(c * 32) * 64 + t * 16 + l16];
                    #pragma unroll
                    for (int j = 0; j < 16; ++j) bv[j] = vp[ka_idx(j, half) * 64];
                    o[t] = __builtin_amdgcn_wmma_f32_16x16x32_f16(
                               false, ap, false, bv, (short)0, o[t], false, false);
                }
            }
        }

        // Hand off buffers: DMA for tile it+1 must be done before next round
        __builtin_amdgcn_s_wait_tensorcnt(0);
        __syncthreads();
    }

    // ---- epilogue: normalize and store f32 ----
    float* ob = out + (size_t)b * TT * HS;
    #pragma unroll
    for (int t = 0; t < 4; ++t)
        #pragma unroll
        for (int r = 0; r < 8; ++r) {
            const int row = q0 + r + 8 * half;
            ob[(size_t)row * HS + t * 16 + l16] = o[t][r] / l_i[r];
        }
}

extern "C" void kernel_launch(void* const* d_in, const int* in_sizes, int n_in,
                              void* d_out, int out_size, void* d_ws, size_t ws_size,
                              hipStream_t stream) {
    const float* x  = (const float*)d_in[0];
    const float* Wq = (const float*)d_in[1];
    const float* Wk = (const float*)d_in[2];
    const float* Wv = (const float*)d_in[3];

    _Float16* Qh = (_Float16*)d_ws;            // 3 x BT x HS f16 = 6 MB scratch
    _Float16* Kh = Qh + (size_t)BT * HS;
    _Float16* Vh = Kh + (size_t)BT * HS;

    proj_qkv_kernel<<<dim3(BT / 16, 3), 128, 0, stream>>>(x, Wq, Wk, Wv, Qh, Kh, Vh);
    flash_attn_kernel<<<dim3(TT / 64, BB), 128, 0, stream>>>(Qh, Kh, Vh, (float*)d_out);
}
